// PPIModel_41858751267052
// MI455X (gfx1250) — matrix operand order repaired and checked by reading, not compiled, not run on hardware
//
#include <hip/hip_runtime.h>
#include <hip/hip_bf16.h>
#include <cstdint>
#include <cstddef>

typedef __attribute__((ext_vector_type(16))) _Float16 v16h;
typedef __attribute__((ext_vector_type(8)))  float    v8f;

static __device__ inline v8f vzero8() {
    v8f v;
    for (int i = 0; i < 8; ++i) v[i] = 0.f;
    return v;
}

// ---------------------------------------------------------------------------
// WMMA GEMM: C[M x (16*NTILES)] = A[M x K] (fp32, cvt->f16) * Bfrag + bias
// Bfrag is pre-laid-out in per-lane fragment order (see make_bfrag_kernel).
// One wave -> one 16-row M tile x all N tiles. 8 waves / block.
// ---------------------------------------------------------------------------
template <int NTILES>
__global__ __launch_bounds__(256) void wmma_gemm_kernel(
    const float* __restrict__ A, const _Float16* __restrict__ Bfrag,
    const float* __restrict__ bias, float* __restrict__ C,
    int M, int K, int relu)
{
    const int lane  = threadIdx.x & 31;
    const int wave  = threadIdx.x >> 5;
    const int row0  = (blockIdx.x * 8 + wave) * 16;
    if (row0 >= M) return;                      // wave-uniform exit

    const int  arow   = row0 + (lane & 15);
    const int  khalf  = (lane >> 4) * 8;        // lanes 0-15: K 0-7/16-23 ; 16-31: K 8-15/24-31
    const int  ksteps = K >> 5;
    const bool rowok  = (arow < M);
    const float* arowp = A + (size_t)arow * K;

    v8f acc[NTILES];
    for (int n = 0; n < NTILES; ++n) acc[n] = vzero8();

    for (int ks = 0; ks < ksteps; ++ks) {
        const int kbase = ks * 32 + khalf;
        v16h a;
        if (rowok) {
            const float4 x0 = *(const float4*)(arowp + kbase + 0);
            const float4 x1 = *(const float4*)(arowp + kbase + 4);
            const float4 y0 = *(const float4*)(arowp + kbase + 16);
            const float4 y1 = *(const float4*)(arowp + kbase + 20);
            a[0]  = (_Float16)x0.x; a[1]  = (_Float16)x0.y; a[2]  = (_Float16)x0.z; a[3]  = (_Float16)x0.w;
            a[4]  = (_Float16)x1.x; a[5]  = (_Float16)x1.y; a[6]  = (_Float16)x1.z; a[7]  = (_Float16)x1.w;
            a[8]  = (_Float16)y0.x; a[9]  = (_Float16)y0.y; a[10] = (_Float16)y0.z; a[11] = (_Float16)y0.w;
            a[12] = (_Float16)y1.x; a[13] = (_Float16)y1.y; a[14] = (_Float16)y1.z; a[15] = (_Float16)y1.w;
            if (ks + 1 < ksteps)
                __builtin_prefetch(arowp + kbase + 32, 0, 0);   // global_prefetch_b8
        } else {
            for (int j = 0; j < 16; ++j) a[j] = (_Float16)0.f;
        }
        // control flow reconverged here: EXEC all-ones for WMMA
        for (int n = 0; n < NTILES; ++n) {
            const v16h b = *(const v16h*)(Bfrag + ((((size_t)n * ksteps) + ks) * 32 + lane) * 16);
            acc[n] = __builtin_amdgcn_wmma_f32_16x16x32_f16(
                         false, a, false, b, (short)0, acc[n], false, false);
        }
    }

    // C/D layout: VGPR r -> M=row0+r (lanes 0-15) / M=row0+8+r (lanes 16-31), N=lane&15
    const int N    = NTILES * 16;
    const int crow = row0 + (lane >> 4) * 8;
    const int ccol = lane & 15;
    for (int n = 0; n < NTILES; ++n) {
        const float bv = bias ? bias[n * 16 + ccol] : 0.f;
        for (int r = 0; r < 8; ++r) {
            const int rr = crow + r;
            if (rr < M) {
                float v = acc[n][r] + bv;
                if (relu) v = fmaxf(v, 0.f);
                C[(size_t)rr * N + n * 16 + ccol] = v;
            }
        }
    }
}

// Convert fp32 weights [K x N] into per-lane WMMA B fragments:
// lane L (<16): column n*16+L, K = ks*32 + 0-7,16-23 ; lanes 16-31: same cols, K +8.
__global__ void make_bfrag_kernel(const float* __restrict__ W, _Float16* __restrict__ out,
                                  int K, int N, int NTILES)
{
    const int ksteps = K >> 5;
    const int total  = NTILES * ksteps * 32;
    const int idx    = blockIdx.x * blockDim.x + threadIdx.x;
    if (idx >= total) return;
    const int lane = idx & 31;
    const int ks   = (idx >> 5) % ksteps;
    const int n    = idx / (ksteps * 32);
    const int col  = n * 16 + (lane & 15);
    const int k0   = ks * 32 + (lane >> 4) * 8;
    _Float16* o = out + (size_t)idx * 16;
    for (int j = 0; j < 16; ++j) o[j] = (_Float16)W[(size_t)(k0 + j + (j >= 8 ? 8 : 0)) * N + col];
}

// ---------------------------------------------------------------------------
// Graph / elementwise kernels
// ---------------------------------------------------------------------------
__global__ void fill_kernel(float* __restrict__ p, float val, long long n)
{
    const long long i = (long long)blockIdx.x * blockDim.x + threadIdx.x;
    if (i < n) p[i] = val;
}

__global__ void degree_kernel(const int* __restrict__ dst, float* __restrict__ deg, int E)
{
    const int e = blockIdx.x * blockDim.x + threadIdx.x;
    if (e < E) atomicAdd(&deg[dst[e]], 1.0f);
}

__global__ void rsqrt_kernel(float* __restrict__ p, int n)
{
    const int i = blockIdx.x * blockDim.x + threadIdx.x;
    if (i < n) p[i] = rsqrtf(p[i]);
}

// agg[dst] += h[src] * coef   (coef = dinv[src]*dinv[dst] for GCN, 1 for GIN)
// 16 threads per edge, float4 per thread over 64 features.
__global__ void scatter_kernel(const float* __restrict__ h, const int* __restrict__ src,
                               const int* __restrict__ dst, const float* __restrict__ dinv,
                               float* __restrict__ agg, int E, int use_coef)
{
    const long long tid = (long long)blockIdx.x * blockDim.x + threadIdx.x;
    if (tid >= (long long)E * 16) return;
    const int e = (int)(tid >> 4);
    const int c = (int)(tid & 15) * 4;
    const int s = src[e], d = dst[e];
    const float coef = use_coef ? dinv[s] * dinv[d] : 1.0f;
    const float4 v = *(const float4*)(h + (size_t)s * 64 + c);
    float* o = agg + (size_t)d * 64 + c;
    atomicAdd(o + 0, v.x * coef);
    atomicAdd(o + 1, v.y * coef);
    atomicAdd(o + 2, v.z * coef);
    atomicAdd(o + 3, v.w * coef);
}

// out = relu(instnorm_row(agg + dinv^2 * h + bias)), C = 64. One wave per row.
__global__ __launch_bounds__(256) void gcn_post_kernel(
    const float* __restrict__ agg, const float* __restrict__ h,
    const float* __restrict__ dinv, const float* __restrict__ bias,
    float* __restrict__ out, int M)
{
    const int lane = threadIdx.x & 31;
    const int wave = threadIdx.x >> 5;
    const int row  = blockIdx.x * 8 + wave;
    if (row >= M) return;
    const float d2 = dinv[row] * dinv[row];
    const size_t base = (size_t)row * 64;
    float v0 = agg[base + lane]      + d2 * h[base + lane]      + bias[lane];
    float v1 = agg[base + 32 + lane] + d2 * h[base + 32 + lane] + bias[32 + lane];

    float s = v0 + v1;
    for (int off = 16; off >= 1; off >>= 1) s += __shfl_xor(s, off, 32);
    const float mean = s * (1.0f / 64.0f);

    float q = (v0 - mean) * (v0 - mean) + (v1 - mean) * (v1 - mean);
    for (int off = 16; off >= 1; off >>= 1) q += __shfl_xor(q, off, 32);
    const float inv = rsqrtf(q * (1.0f / 64.0f) + 1e-5f);

    out[base + lane]      = fmaxf((v0 - mean) * inv, 0.f);
    out[base + 32 + lane] = fmaxf((v1 - mean) * inv, 0.f);
}

__global__ void add_kernel(float* __restrict__ acc, const float* __restrict__ x, long long n)
{
    const long long i = (long long)blockIdx.x * blockDim.x + threadIdx.x;
    if (i < n) acc[i] += x[i];
}

// Column sum / sumsq for batch norm (stats[0..C)=sum, stats[C..2C)=sumsq).
__global__ void bn_stats_kernel(const float* __restrict__ x, float* __restrict__ stats,
                                int M, int C)
{
    const int t   = threadIdx.x;
    const int c   = t % C;
    const int tpc = blockDim.x / C;
    const int r0  = blockIdx.x * tpc + t / C;
    float s = 0.f, ss = 0.f;
    for (int r = r0; r < M; r += gridDim.x * tpc) {
        const float v = x[(size_t)r * C + c];
        s += v; ss += v * v;
    }
    atomicAdd(&stats[c], s);
    atomicAdd(&stats[C + c], ss);
}

__global__ void bn_finalize_kernel(float* __restrict__ stats, const float* __restrict__ g,
                                   const float* __restrict__ b, float invM, int C)
{
    const int c = threadIdx.x;
    if (c < C) {
        const float mean = stats[c] * invM;
        const float var  = stats[C + c] * invM - mean * mean;
        const float a    = g[c] * rsqrtf(var + 1e-5f);
        stats[c]     = a;
        stats[C + c] = b[c] - mean * a;
    }
}

__global__ void bn_apply_relu_kernel(const float* __restrict__ x, const float* __restrict__ stats,
                                     float* __restrict__ out, long long n, int C)
{
    const long long i = (long long)blockIdx.x * blockDim.x + threadIdx.x;
    if (i < n) {
        const int c = (int)(i % C);
        out[i] = fmaxf(x[i] * stats[c] + stats[C + c], 0.f);
    }
}

__global__ void pool_kernel(const float* __restrict__ h, const int* __restrict__ batch,
                            float* __restrict__ pooled, int M)
{
    const long long tid = (long long)blockIdx.x * blockDim.x + threadIdx.x;
    if (tid >= (long long)M * 32) return;
    const int i = (int)(tid >> 5);
    const int c = (int)(tid & 31);
    atomicAdd(&pooled[(size_t)batch[i] * 32 + c], h[(size_t)i * 32 + c]);
}

__global__ void fc_kernel(const float* __restrict__ pooled, const float* __restrict__ W,
                          const float* __restrict__ b, float* __restrict__ out, int G)
{
    const int g = threadIdx.x;
    if (g < G) {
        float s = b[0];
        for (int c = 0; c < 32; ++c) s += pooled[g * 32 + c] * W[c];
        out[g] = s;
    }
}

// ---------------------------------------------------------------------------
extern "C" void kernel_launch(void* const* d_in, const int* in_sizes, int n_in,
                              void* d_out, int out_size, void* d_ws, size_t ws_size,
                              hipStream_t stream)
{
    (void)n_in; (void)ws_size;

    const float* x    = (const float*)d_in[0];
    const int*   ei   = (const int*)d_in[1];
    const int*   batch= (const int*)d_in[2];
    const float* W1   = (const float*)d_in[3];
    const float* b1   = (const float*)d_in[4];
    const float* W2   = (const float*)d_in[5];
    const float* b2   = (const float*)d_in[6];
    const float* l1W  = (const float*)d_in[7];
    const float* l1b  = (const float*)d_in[8];
    const float* l2W  = (const float*)d_in[9];
    const float* l2b  = (const float*)d_in[10];
    const float* bn1g = (const float*)d_in[11];
    const float* bn1b = (const float*)d_in[12];
    const float* l3W  = (const float*)d_in[13];
    const float* l3b  = (const float*)d_in[14];
    const float* bn2g = (const float*)d_in[15];
    const float* bn2b = (const float*)d_in[16];
    const float* fcW  = (const float*)d_in[17];
    const float* fcb  = (const float*)d_in[18];

    const int M  = in_sizes[2];            // 50000 nodes
    const int E  = in_sizes[1] / 2;        // 800000 edges
    const int K1 = in_sizes[0] / M;        // 1024
    const int G  = out_size;               // 64 graphs
    const int*   src = ei;
    const int*   dst = ei + E;

    // -------- workspace carve-up --------
    char* base = (char*)d_ws;
    size_t off = 0;
    auto alloc = [&](size_t bytes) -> char* {
        char* p = base + off;
        off = (off + bytes + 255) & ~(size_t)255;
        return p;
    };
    float*    bufA   = (float*)alloc((size_t)M * 64 * 4);
    float*    bufB   = (float*)alloc((size_t)M * 64 * 4);
    float*    bufC   = (float*)alloc((size_t)M * 64 * 4);
    float*    dinv   = (float*)alloc((size_t)M * 4);
    _Float16* frag1  = (_Float16*)alloc((size_t)4 * (K1 / 32) * 32 * 16 * 2);
    _Float16* frag2  = (_Float16*)alloc((size_t)4 * 2 * 32 * 16 * 2);
    _Float16* fragL1 = (_Float16*)alloc((size_t)4 * 2 * 32 * 16 * 2);
    _Float16* fragL2 = (_Float16*)alloc((size_t)4 * 2 * 32 * 16 * 2);
    _Float16* fragL3 = (_Float16*)alloc((size_t)2 * 2 * 32 * 16 * 2);
    float*    stats  = (float*)alloc(2 * 64 * 4);
    float*    pooled = (float*)alloc((size_t)G * 32 * 4);

    const int  TB = 256;
    auto nblk  = [&](long long n) { return (int)((n + TB - 1) / TB); };
    const long long MC  = (long long)M * 64;
    const long long MC2 = (long long)M * 32;
    const int gemmBlocks = ((M + 15) / 16 + 7) / 8;
    const int postBlocks = (M + 7) / 8;
    const int scatBlocks = nblk((long long)E * 16);

    // -------- degree / dinv --------
    fill_kernel<<<nblk(M), TB, 0, stream>>>(dinv, 1.0f, M);          // self loop
    degree_kernel<<<nblk(E), TB, 0, stream>>>(dst, dinv, E);
    rsqrt_kernel<<<nblk(M), TB, 0, stream>>>(dinv, M);

    // -------- weight fragments --------
    make_bfrag_kernel<<<nblk(4 * (K1 / 32) * 32), TB, 0, stream>>>(W1,  frag1,  K1, 64, 4);
    make_bfrag_kernel<<<1, TB, 0, stream>>>(W2,  frag2,  64, 64, 4);
    make_bfrag_kernel<<<1, TB, 0, stream>>>(l1W, fragL1, 64, 64, 4);
    make_bfrag_kernel<<<1, TB, 0, stream>>>(l2W, fragL2, 64, 64, 4);
    make_bfrag_kernel<<<1, TB, 0, stream>>>(l3W, fragL3, 64, 32, 2);

    // -------- GCN layer 1 --------
    wmma_gemm_kernel<4><<<gemmBlocks, TB, 0, stream>>>(x, frag1, nullptr, bufA, M, K1, 0);
    fill_kernel<<<nblk(MC), TB, 0, stream>>>(bufB, 0.f, MC);
    scatter_kernel<<<scatBlocks, TB, 0, stream>>>(bufA, src, dst, dinv, bufB, E, 1);
    gcn_post_kernel<<<postBlocks, TB, 0, stream>>>(bufB, bufA, dinv, b1, bufC, M);

    // -------- GCN layer 2 --------
    wmma_gemm_kernel<4><<<gemmBlocks, TB, 0, stream>>>(bufC, frag2, nullptr, bufA, M, 64, 0);
    fill_kernel<<<nblk(MC), TB, 0, stream>>>(bufB, 0.f, MC);
    scatter_kernel<<<scatBlocks, TB, 0, stream>>>(bufA, src, dst, dinv, bufB, E, 1);
    gcn_post_kernel<<<postBlocks, TB, 0, stream>>>(bufB, bufA, dinv, b2, bufC, M);

    // -------- GIN conv 1 + MLP1 + BN1 --------
    fill_kernel<<<nblk(MC), TB, 0, stream>>>(bufB, 0.f, MC);
    scatter_kernel<<<scatBlocks, TB, 0, stream>>>(bufC, src, dst, dinv, bufB, E, 0);
    add_kernel<<<nblk(MC), TB, 0, stream>>>(bufB, bufC, MC);
    wmma_gemm_kernel<4><<<gemmBlocks, TB, 0, stream>>>(bufB, fragL1, l1b, bufA, M, 64, 1);
    wmma_gemm_kernel<4><<<gemmBlocks, TB, 0, stream>>>(bufA, fragL2, l2b, bufC, M, 64, 0);
    fill_kernel<<<1, TB, 0, stream>>>(stats, 0.f, 128);
    bn_stats_kernel<<<256, TB, 0, stream>>>(bufC, stats, M, 64);
    bn_finalize_kernel<<<1, 64, 0, stream>>>(stats, bn1g, bn1b, 1.0f / (float)M, 64);
    bn_apply_relu_kernel<<<nblk(MC), TB, 0, stream>>>(bufC, stats, bufA, MC, 64);

    // -------- GIN conv 2 + MLP2 + BN2 --------
    fill_kernel<<<nblk(MC), TB, 0, stream>>>(bufB, 0.f, MC);
    scatter_kernel<<<scatBlocks, TB, 0, stream>>>(bufA, src, dst, dinv, bufB, E, 0);
    add_kernel<<<nblk(MC), TB, 0, stream>>>(bufB, bufA, MC);
    wmma_gemm_kernel<2><<<gemmBlocks, TB, 0, stream>>>(bufB, fragL3, l3b, bufC, M, 64, 0);
    fill_kernel<<<1, TB, 0, stream>>>(stats, 0.f, 64);
    bn_stats_kernel<<<256, TB, 0, stream>>>(bufC, stats, M, 32);
    bn_finalize_kernel<<<1, 32, 0, stream>>>(stats, bn2g, bn2b, 1.0f / (float)M, 32);
    bn_apply_relu_kernel<<<nblk(MC2), TB, 0, stream>>>(bufC, stats, bufA, MC2, 32);

    // -------- pool + FC --------
    fill_kernel<<<1, TB, 0, stream>>>(pooled, 0.f, (long long)G * 32);
    pool_kernel<<<nblk(MC2), TB, 0, stream>>>(bufA, batch, pooled, M);
    fc_kernel<<<1, 64, 0, stream>>>(pooled, fcW, fcb, (float*)d_out, G);
}
